// SelfAttention_58085137711838
// MI455X (gfx1250) — matrix operand order repaired and checked by reading, not compiled, not run on hardware
//
#include <hip/hip_runtime.h>
#include <hip/hip_bf16.h>

// ---------------------------------------------------------------------------
// CDNA5 (gfx1250) self-attention forward, bf16 WMMA + async-LDS pipeline:
//   Q = X Wq + bq ; K = X Wk + bk ; Vt = (X Wv + bv)^T    (proj kernel)
//   P = softmax(Q K^T / sqrt(DK))                          (gemm_nt + softmax)
//   O = P V  (== P @ Vt^T)                                 (gemm_nt, f32 out)
// B=4, S=2048, D=DK=DV=1024.
// ---------------------------------------------------------------------------

typedef __attribute__((ext_vector_type(16))) __bf16 v16bf;
typedef __attribute__((ext_vector_type(8)))  float  v8f;
typedef int v4i __attribute__((vector_size(16)));   // matches builtin param type

union FragB { unsigned int u[8]; v16bf v; };
union FragC { float        f[8]; v8f   v; };

__device__ __forceinline__ unsigned short f2bf(float x) {
    unsigned int u = __float_as_uint(x);
    u = u + 0x7FFFu + ((u >> 16) & 1u);          // round-to-nearest-even
    return (unsigned short)(u >> 16);
}
__device__ __forceinline__ unsigned int pack2_bf16(float lo, float hi) {
    return (unsigned int)f2bf(lo) | ((unsigned int)f2bf(hi) << 16);
}
__device__ __forceinline__ float bf2f(unsigned short h) {
    return __uint_as_float(((unsigned int)h) << 16);
}

// CDNA5 16-bit A-matrix (16x32) packed-pair index for VGPR v, lane-half h
__device__ __forceinline__ int apair(int v, int h) {
    return (v & 3) + h * 4 + ((v >> 2) << 3);
}
// CDNA5 16-bit B-matrix (32x16) packed-pair index for VGPR v, lane-half h
__device__ __forceinline__ int bpair(int v, int h) {
    return (v & 3) + h * 4 + ((v >> 2) << 3);
}

// ---- gfx1250 async global->LDS copy (ASYNCcnt-tracked), with fallback ------
#if defined(__gfx1250__) && __has_builtin(__builtin_amdgcn_global_load_async_to_lds_b128)
#define HAVE_ASYNC_LDS 1
#define ASYNC_COPY16(gsrc, ldst)                                              \
    __builtin_amdgcn_global_load_async_to_lds_b128(                           \
        (__attribute__((address_space(1))) v4i*)(gsrc),                       \
        (__attribute__((address_space(3))) v4i*)(ldst), 0, 0)
#if __has_builtin(__builtin_amdgcn_s_wait_asynccnt)
#define ASYNC_WAIT(n) __builtin_amdgcn_s_wait_asynccnt(n)
#else
#define ASYNC_WAIT(n) asm volatile("s_wait_asynccnt %0" ::"i"(n) : "memory")
#endif
#else
#define HAVE_ASYNC_LDS 0
#define ASYNC_COPY16(gsrc, ldst) (*(uint4*)(ldst) = *(const uint4*)(gsrc))
#define ASYNC_WAIT(n) ((void)0)
#endif

#define BM 128
#define BN 128
#define BK 32
#define NPAIR (BK / 2)   // 16 packed bf16 pairs per tile row

// ---------------------------------------------------------------------------
// Kernel 1: C[M,N] = bf16( X_f32[M,K] @ W_f32[K,N] + bias[N] )
// If trS > 0, output is stored batch-transposed: Out[b][col][s] (ld = trS).
// ---------------------------------------------------------------------------
__global__ __launch_bounds__(256)
void proj_gemm_bf16(const float* __restrict__ X, const float* __restrict__ W,
                    const float* __restrict__ bias, unsigned short* __restrict__ Out,
                    int Kd, int N, int trS) {
    __shared__ unsigned int As[BM][NPAIR + 1];
    __shared__ unsigned int Bs[BN][NPAIR + 1];

    const int tid  = threadIdx.x;
    const int bm   = blockIdx.y, bn = blockIdx.x;
    const int wave = tid >> 5, lane = tid & 31;
    const int wm = wave & 3, wn = wave >> 2;     // 4 x 2 wave grid
    const int lm = lane & 15, half = lane >> 4;

    FragC acc[2][4];
#pragma unroll
    for (int mt = 0; mt < 2; ++mt)
#pragma unroll
        for (int nt = 0; nt < 4; ++nt)
#pragma unroll
            for (int r = 0; r < 8; ++r) acc[mt][nt].f[r] = 0.0f;

    for (int k0 = 0; k0 < Kd; k0 += BK) {
        if (k0 + BK < Kd) {   // hint next tiles into cache (global_prefetch_b8)
            __builtin_prefetch(X + (size_t)(bm * BM + (tid >> 1)) * Kd + k0 + BK, 0, 0);
            __builtin_prefetch(W + (size_t)(k0 + BK + (tid >> 3)) * N + bn * BN, 0, 0);
        }
        // A tile: 128 rows x 16 pairs from X (fp32 -> packed bf16)
#pragma unroll
        for (int i = 0; i < 8; ++i) {
            int idx = tid + 256 * i;             // 0..2047
            int r = idx >> 4, c = idx & 15;
            const float* p = X + (size_t)(bm * BM + r) * Kd + k0 + 2 * c;
            As[r][c] = pack2_bf16(p[0], p[1]);
        }
        // B tile transposed: Bs[n][kpair] from W[k][n]
#pragma unroll
        for (int i = 0; i < 8; ++i) {
            int idx = tid + 256 * i;
            int n = idx & 127, c = idx >> 7;
            const float* p = W + (size_t)(k0 + 2 * c) * N + bn * BN + n;
            Bs[n][c] = pack2_bf16(p[0], p[N]);
        }
        __syncthreads();

        FragB a[2], b[4];
#pragma unroll
        for (int mt = 0; mt < 2; ++mt)
#pragma unroll
            for (int v = 0; v < 8; ++v)
                a[mt].u[v] = As[wm * 32 + mt * 16 + lm][apair(v, half)];
#pragma unroll
        for (int nt = 0; nt < 4; ++nt)
#pragma unroll
            for (int v = 0; v < 8; ++v)
                b[nt].u[v] = Bs[wn * 64 + nt * 16 + lm][bpair(v, half)];

#pragma unroll
        for (int mt = 0; mt < 2; ++mt)
#pragma unroll
            for (int nt = 0; nt < 4; ++nt)
                acc[mt][nt].v = __builtin_amdgcn_wmma_f32_16x16x32_bf16(
                    false, a[mt].v, false, b[nt].v, (short)0, acc[mt][nt].v, false, false);
        __syncthreads();
    }

#pragma unroll
    for (int mt = 0; mt < 2; ++mt)
#pragma unroll
        for (int nt = 0; nt < 4; ++nt)
#pragma unroll
            for (int r = 0; r < 8; ++r) {
                int row = bm * BM + wm * 32 + mt * 16 + half * 8 + r;
                int col = bn * BN + wn * 64 + nt * 16 + lm;
                float val = acc[mt][nt].f[r] + bias[col];
                if (trS > 0) {                   // batch-transposed store
                    int bb = row / trS, s = row - bb * trS;
                    Out[((size_t)bb * N + col) * trS + s] = f2bf(val);
                } else {
                    Out[(size_t)row * N + col] = f2bf(val);
                }
            }
}

// ---------------------------------------------------------------------------
// Kernel 2: NT bf16 GEMM with async double-buffered LDS.
//   C[M,N] = A[M,Kd] @ Bt[N,Kd]^T, batched via blockIdx.z.
// Block tile 256x128, 8 waves (4x2), wave tile 64x64 (16 WMMAs / k-step).
// OUT_F32: write f32 (AV). else: write bf16 * scale (scores).
// ---------------------------------------------------------------------------
#define BM2 256
#define BN2 128

template <bool OUT_F32>
__global__ __launch_bounds__(256)
void gemm_nt_bf16(const unsigned short* __restrict__ A, size_t aBatch,
                  const unsigned short* __restrict__ Bt, size_t bBatch,
                  void* __restrict__ OutP, size_t oBatch,
                  int Kd, int N, float scale) {
    __shared__ unsigned int As2[2][BM2][NPAIR];   // 32 KB (no pad: 16B-aligned rows)
    __shared__ unsigned int Bs2[2][BN2][NPAIR];   // 16 KB

    const int tid  = threadIdx.x;
    const int bm   = blockIdx.y, bn = blockIdx.x;
    const int wave = tid >> 5, lane = tid & 31;
    const int wm = wave & 3, wn = wave >> 2;      // 4 x 2 wave grid
    const int lm = lane & 15, half = lane >> 4;
    const int kp = Kd >> 1;                       // uint pairs per row

    const unsigned int* Au  = (const unsigned int*)(A  + (size_t)blockIdx.z * aBatch);
    const unsigned int* Btu = (const unsigned int*)(Bt + (size_t)blockIdx.z * bBatch);

    FragC acc[4][4];
#pragma unroll
    for (int mt = 0; mt < 4; ++mt)
#pragma unroll
        for (int nt = 0; nt < 4; ++nt)
#pragma unroll
            for (int r = 0; r < 8; ++r) acc[mt][nt].f[r] = 0.0f;

    // --- tile copy helpers (6 x 16B chunks per thread per k-step) ---
    auto copy_tiles = [&](int buf, int kpair0) {
#pragma unroll
        for (int i = 0; i < 4; ++i) {             // A: 256 rows x 4 chunks
            int idx = tid + 256 * i;
            int r = idx >> 2, g = (idx & 3) << 2;
            ASYNC_COPY16(Au + (size_t)(bm * BM2 + r) * kp + kpair0 + g,
                         &As2[buf][r][g]);
        }
#pragma unroll
        for (int i = 0; i < 2; ++i) {             // B: 128 rows x 4 chunks
            int idx = tid + 256 * i;
            int r = idx >> 2, g = (idx & 3) << 2;
            ASYNC_COPY16(Btu + (size_t)(bn * BN2 + r) * kp + kpair0 + g,
                         &Bs2[buf][r][g]);
        }
    };

    const int TK = Kd / BK;
    copy_tiles(0, 0);
    int buf = 0;
    for (int t = 0; t < TK; ++t) {
        if (t + 1 < TK) {
            copy_tiles(buf ^ 1, (t + 1) * NPAIR); // prefetch next tile
            ASYNC_WAIT(6);                        // 6 newest in flight = next tile
        } else {
            ASYNC_WAIT(0);
        }
        __syncthreads();                          // all lanes' current tile landed

        FragB a[4], b[4];
#pragma unroll
        for (int mt = 0; mt < 4; ++mt)
#pragma unroll
            for (int v = 0; v < 8; ++v)
                a[mt].u[v] = As2[buf][wm * 64 + mt * 16 + lm][apair(v, half)];
#pragma unroll
        for (int nt = 0; nt < 4; ++nt)
#pragma unroll
            for (int v = 0; v < 8; ++v)
                b[nt].u[v] = Bs2[buf][wn * 64 + nt * 16 + lm][bpair(v, half)];

#pragma unroll
        for (int mt = 0; mt < 4; ++mt)
#pragma unroll
            for (int nt = 0; nt < 4; ++nt)
                acc[mt][nt].v = __builtin_amdgcn_wmma_f32_16x16x32_bf16(
                    false, a[mt].v, false, b[nt].v, (short)0, acc[mt][nt].v, false, false);

        __syncthreads();                          // done reading buf before reuse
        buf ^= 1;
    }

#pragma unroll
    for (int mt = 0; mt < 4; ++mt)
#pragma unroll
        for (int nt = 0; nt < 4; ++nt)
#pragma unroll
            for (int r = 0; r < 8; ++r) {
                int row = bm * BM2 + wm * 64 + mt * 16 + half * 8 + r;
                int col = bn * BN2 + wn * 64 + nt * 16 + lm;
                if (OUT_F32) {
                    float* out = (float*)OutP + (size_t)blockIdx.z * oBatch;
                    out[(size_t)row * N + col] = acc[mt][nt].f[r];
                } else {
                    unsigned short* out = (unsigned short*)OutP + (size_t)blockIdx.z * oBatch;
                    out[(size_t)row * N + col] = f2bf(acc[mt][nt].f[r] * scale);
                }
            }
}

// ---------------------------------------------------------------------------
// Kernel 3: in-place row softmax over bf16 scores. One block per row, S=2048.
// ---------------------------------------------------------------------------
__global__ __launch_bounds__(256)
void softmax_rows_bf16(unsigned short* __restrict__ P, int S) {
    __shared__ float red[256];
    const int tid = threadIdx.x;
    unsigned short* p = P + (size_t)blockIdx.x * S;

    float v[8];
    float mx = -3.0e38f;
#pragma unroll
    for (int i = 0; i < 8; ++i) {
        v[i] = bf2f(p[tid + 256 * i]);
        mx = fmaxf(mx, v[i]);
    }
    red[tid] = mx; __syncthreads();
#pragma unroll
    for (int s = 128; s > 0; s >>= 1) {
        if (tid < s) red[tid] = fmaxf(red[tid], red[tid + s]);
        __syncthreads();
    }
    mx = red[0]; __syncthreads();

    float sum = 0.0f;
#pragma unroll
    for (int i = 0; i < 8; ++i) {
        v[i] = __expf(v[i] - mx);
        sum += v[i];
    }
    red[tid] = sum; __syncthreads();
#pragma unroll
    for (int s = 128; s > 0; s >>= 1) {
        if (tid < s) red[tid] += red[tid + s];
        __syncthreads();
    }
    const float inv = 1.0f / red[0];
#pragma unroll
    for (int i = 0; i < 8; ++i)
        p[tid + 256 * i] = f2bf(v[i] * inv);
}

// ---------------------------------------------------------------------------
extern "C" void kernel_launch(void* const* d_in, const int* in_sizes, int n_in,
                              void* d_out, int out_size, void* d_ws, size_t ws_size,
                              hipStream_t stream) {
    const int B = 4, S = 2048, D = 1024, DK = 1024, DV = 1024;
    const float* X  = (const float*)d_in[0];
    const float* Wq = (const float*)d_in[1];
    const float* Wk = (const float*)d_in[2];
    const float* Wv = (const float*)d_in[3];
    const float* bq = (const float*)d_in[4];
    const float* bk = (const float*)d_in[5];
    const float* bv = (const float*)d_in[6];
    float* Out = (float*)d_out;

    char* ws = (char*)d_ws;
    const size_t qkv_bytes = (size_t)B * S * DK * sizeof(unsigned short); // 16 MB
    unsigned short* Q  = (unsigned short*)(ws);
    unsigned short* K  = (unsigned short*)(ws + qkv_bytes);
    unsigned short* Vt = (unsigned short*)(ws + 2 * qkv_bytes);  // [B][DV][S]
    unsigned short* P  = (unsigned short*)(ws + 3 * qkv_bytes);  // 32 MB

    dim3 blk(256);

    // 1) QKV projections (M = B*S = 8192 rows). V stored batch-transposed.
    dim3 gproj(DK / BN, (B * S) / BM);
    proj_gemm_bf16<<<gproj, blk, 0, stream>>>(X, Wq, bq, Q,  D, DK, 0);
    proj_gemm_bf16<<<gproj, blk, 0, stream>>>(X, Wk, bk, K,  D, DK, 0);
    proj_gemm_bf16<<<gproj, blk, 0, stream>>>(X, Wv, bv, Vt, D, DV, S);

    // 2) Scores = Q K^T * 1/sqrt(DK)  -> bf16 P
    dim3 gsc(S / BN2, S / BM2, B);
    gemm_nt_bf16<false><<<gsc, blk, 0, stream>>>(
        Q, (size_t)S * DK, K, (size_t)S * DK,
        (void*)P, (size_t)S * S, DK, S, 0.03125f);

    // 3) Row softmax (in place)
    softmax_rows_bf16<<<B * S, blk, 0, stream>>>(P, S);

    // 4) O = P @ Vt^T  (f32 out)
    dim3 gav(DV / BN2, S / BM2, B);
    gemm_nt_bf16<true><<<gav, blk, 0, stream>>>(
        P, (size_t)S * S, Vt, (size_t)DV * S,
        (void*)Out, (size_t)S * DV, S, DV, 1.0f);
}
